// MS_MSA_C_88003879895540
// MI455X (gfx1250) — compile-verified, matrix operand order checked by reading, not run on hardware
//
#include <hip/hip_runtime.h>
#include <hip/hip_bf16.h>
#include <math.h>

typedef __bf16 bf16_t;
typedef bf16_t v16bf __attribute__((ext_vector_type(16)));
typedef bf16_t v8bf  __attribute__((ext_vector_type(8)));
typedef float  v8f   __attribute__((ext_vector_type(8)));
typedef unsigned int v4u __attribute__((ext_vector_type(4)));
typedef int v8i __attribute__((ext_vector_type(8)));
typedef int v4i __attribute__((ext_vector_type(4)));

#define CB  2
#define CH  128
#define CW  128
#define CDIM 256
#define CHEADS 16
#define CDH 64
#define CINNER 1024
#define CN  16384   // CH*CW

#define GLOBAL_AS __attribute__((address_space(1)))
#define LDS_AS    __attribute__((address_space(3)))

typedef GLOBAL_AS v4i* gptr_v4i;   // pointer to v4i in global addrspace(1)
typedef LDS_AS    v4i* lptr_v4i;   // pointer to v4i in LDS addrspace(3)

// --- gfx1250 async / TDM feature detection (compile-safe) -------------------
#if defined(__has_builtin)
#  if __has_builtin(__builtin_amdgcn_global_load_async_to_lds_b128)
#    define HAVE_ASYNC_LDS 1
#  endif
#  if __has_builtin(__builtin_amdgcn_tensor_load_to_lds) && __has_builtin(__builtin_amdgcn_s_wait_tensorcnt)
#    define HAVE_TDM 1
#  endif
#endif
#ifndef HAVE_ASYNC_LDS
#  define HAVE_ASYNC_LDS 0
#endif
#ifndef HAVE_TDM
#  define HAVE_TDM 0
#endif

static __device__ __forceinline__ void wait_async0() {
#if defined(__has_builtin)
#  if __has_builtin(__builtin_amdgcn_s_wait_asynccnt)
  __builtin_amdgcn_s_wait_asynccnt(0);
#  else
  asm volatile("s_wait_asynccnt 0x0" ::: "memory");
#  endif
#else
  asm volatile("s_wait_asynccnt 0x0" ::: "memory");
#endif
}

#if HAVE_ASYNC_LDS
static __device__ __forceinline__ void async_b128(const bf16_t* g, bf16_t* l) {
  __builtin_amdgcn_global_load_async_to_lds_b128(
      (gptr_v4i)(void*)g, (lptr_v4i)(void*)l, 0, 0);
}
#endif

static __device__ __forceinline__ v8f wmma_bf16(v16bf a, v16bf b, v8f c) {
  // D = A(16x32) * B(32x16) + C(16x16), f32 accumulate
  return __builtin_amdgcn_wmma_f32_16x16x32_bf16(false, a, false, b, (short)0, c, false, false);
}

static __device__ __forceinline__ v8f vzero8() {
  v8f z;
#pragma unroll
  for (int i = 0; i < 8; ++i) z[i] = 0.0f;
  return z;
}

static __device__ __forceinline__ float gelu_exact(float x) {
  return 0.5f * x * (1.0f + erff(x * 0.70710678118654752f));
}

#if HAVE_TDM
// Issue one TDM 2D tile load: 32 (k, contiguous 2B elems) x 64 (rows, stride
// 1024 elems) from Wb panel into LDS (row-major 64x32, 4KB). One issue per wave.
static __device__ __forceinline__ void tdm_load_panel(const bf16_t* Wb, int colBlock, int kb,
                                                      bf16_t* ldsDst) {
  unsigned lds_off = (unsigned)(unsigned long long)(LDS_AS bf16_t*)ldsDst;
  unsigned long long ga = (unsigned long long)(Wb + (unsigned long long)colBlock * CINNER + kb);
  v4u g0;
  g0[0] = 1u;                                   // count=1, user descriptor
  g0[1] = lds_off;                              // lds_addr (bytes)
  g0[2] = (unsigned)ga;                         // global_addr[31:0]
  g0[3] = (unsigned)((ga >> 32) & 0x1FFFFFFu) | 0x80000000u; // addr[56:32] | type=2
  v8i g1;
  g1[0] = 0x00010000;        // workgroup_mask=0, data_size=1 (2 bytes)
  g1[1] = (int)(1024u << 16);// tensor_dim0[15:0]=1024 (atomic_barrier_addr=0)
  g1[2] = (int)(256u << 16); // tensor_dim0[31:16]=0, tensor_dim1[15:0]=256
  g1[3] = (int)(32u << 16);  // tensor_dim1[31:16]=0, tile_dim0=32
  g1[4] = 64;                // tile_dim1=64, tile_dim2=0
  g1[5] = 1024;              // tensor_dim0_stride[31:0]=1024
  g1[6] = 0;                 // stride hi / dim1_stride lo
  g1[7] = 0;
  v4i g2; g2[0] = 0; g2[1] = 0; g2[2] = 0; g2[3] = 0;
  v4i g3 = g2;
  v8i g4; // unused trailing SGPR group in the 6-arg builtin: zero
#pragma unroll
  for (int i = 0; i < 8; ++i) g4[i] = 0;
  __builtin_amdgcn_tensor_load_to_lds(g0, g1, g2, g3, g4, 0);
}
#endif

// ---------------------------------------------------------------------------
// Kernel 0: f32 -> bf16 elementwise convert (proj_w)
// ---------------------------------------------------------------------------
__global__ void cvt_bf16_kernel(const float* __restrict__ src, bf16_t* __restrict__ dst, int nElem) {
  int i = blockIdx.x * 256 + threadIdx.x;
  if (i < nElem) dst[i] = (bf16_t)src[i];
}

// ---------------------------------------------------------------------------
// Kernel 1: grouped conv3x3 (256 groups, 4 out-ch per in-ch) + BN for q,k,v.
// One block per pixel; thread = input channel; computes 4 out channels x 3 paths.
// Writes: qh/kh as bf16 [b, co(=head*64+d), n]   (row-major in n)
//         v_row as bf16 [b, n, co]               (row-major in co)
// ---------------------------------------------------------------------------
__global__ __launch_bounds__(256) void qkv_conv_bn_kernel(
    const float* __restrict__ x_in,
    const float* __restrict__ wq, const float* __restrict__ wk, const float* __restrict__ wv,
    const float* __restrict__ bnq_g, const float* __restrict__ bnq_b,
    const float* __restrict__ bnq_m, const float* __restrict__ bnq_v,
    const float* __restrict__ bnk_g, const float* __restrict__ bnk_b,
    const float* __restrict__ bnk_m, const float* __restrict__ bnk_v,
    const float* __restrict__ bnv_g, const float* __restrict__ bnv_b,
    const float* __restrict__ bnv_m, const float* __restrict__ bnv_v,
    bf16_t* __restrict__ qh, bf16_t* __restrict__ kh, bf16_t* __restrict__ vrow)
{
  const int x = blockIdx.x, y = blockIdx.y, b = blockIdx.z;
  const int cin = threadIdx.x;
  float nb[9];
#pragma unroll
  for (int dy = 0; dy < 3; ++dy)
#pragma unroll
    for (int dx = 0; dx < 3; ++dx) {
      int yy = y + dy - 1, xx = x + dx - 1;
      float val = 0.0f;
      if (yy >= 0 && yy < CH && xx >= 0 && xx < CW)
        val = x_in[(((size_t)b * CH + yy) * CW + xx) * CDIM + cin];
      nb[dy * 3 + dx] = val;
    }
  const int n = y * CW + x;
#pragma unroll
  for (int j = 0; j < 4; ++j) {
    const int co = cin * 4 + j;
    float aq = 0.0f, ak = 0.0f, av = 0.0f;
#pragma unroll
    for (int t = 0; t < 9; ++t) {
      aq = fmaf(nb[t], wq[co * 9 + t], aq);
      ak = fmaf(nb[t], wk[co * 9 + t], ak);
      av = fmaf(nb[t], wv[co * 9 + t], av);
    }
    float sq = bnq_g[co] * rsqrtf(bnq_v[co] + 1e-5f);
    float sk = bnk_g[co] * rsqrtf(bnk_v[co] + 1e-5f);
    float sv = bnv_g[co] * rsqrtf(bnv_v[co] + 1e-5f);
    float oq = (aq - bnq_m[co]) * sq + bnq_b[co];
    float ok = (ak - bnk_m[co]) * sk + bnk_b[co];
    float ov = (av - bnv_m[co]) * sv + bnv_b[co];
    size_t ci = ((size_t)b * CINNER + co) * (size_t)CN + n;
    qh[ci] = (bf16_t)oq;
    kh[ci] = (bf16_t)ok;
    vrow[((size_t)b * CN + n) * CINNER + co] = (bf16_t)ov;
  }
}

// ---------------------------------------------------------------------------
// Kernel 2: reciprocal L2 norm of each length-16384 row.
// ---------------------------------------------------------------------------
__global__ __launch_bounds__(256) void rownorm_kernel(const bf16_t* __restrict__ X,
                                                      float* __restrict__ R) {
  const int row = blockIdx.x;
  const int tid = threadIdx.x;
  const bf16_t* p = X + (size_t)row * CN;
  float s = 0.0f;
#pragma unroll
  for (int pass = 0; pass < 4; ++pass) {
    v16bf vv = *(const v16bf*)(p + pass * 4096 + tid * 16);
#pragma unroll
    for (int i = 0; i < 16; ++i) {
      float f = (float)vv[i];
      s = fmaf(f, f, s);
    }
  }
  __shared__ float sm[256];
  sm[tid] = s;
  __syncthreads();
  for (int off = 128; off > 0; off >>= 1) {
    if (tid < off) sm[tid] += sm[tid + off];
    __syncthreads();
  }
  if (tid == 0) {
    float nrm = sqrtf(sm[0]);
    R[row] = 1.0f / fmaxf(nrm, 1e-12f);
  }
}

// ---------------------------------------------------------------------------
// Kernel 3: per-(b,head) logits G = K * Q^T (64x64, contract over n=16384),
// 8 waves split n, 16 WMMA tiles per wave, LDS f32-atomic merge, fused
// norm-scaling + rescale + softmax, bf16 attention out.
// ---------------------------------------------------------------------------
__global__ __launch_bounds__(256) void attn_kernel(
    const bf16_t* __restrict__ qh, const bf16_t* __restrict__ kh,
    const float* __restrict__ rq, const float* __restrict__ rk,
    const float* __restrict__ rescale, bf16_t* __restrict__ attnW)
{
  const int bh = blockIdx.x;
  const int h = bh & (CHEADS - 1);
  const int tid = threadIdx.x;
  const int wave = tid >> 5, lane = tid & 31, idx = lane & 15, half = lane >> 4;

  __shared__ float sG[64 * 64];
  for (int i = tid; i < 4096; i += 256) sG[i] = 0.0f;
  __syncthreads();

  const bf16_t* Kp = kh + (size_t)bh * 64 * CN;
  const bf16_t* Qp = qh + (size_t)bh * 64 * CN;

  v8f acc[4][4];
#pragma unroll
  for (int mt = 0; mt < 4; ++mt)
#pragma unroll
    for (int et = 0; et < 4; ++et) acc[mt][et] = vzero8();

  const int nb0 = wave * 2048;
  for (int nbv = nb0; nbv < nb0 + 2048; nbv += 32) {
    v16bf amat[4], bmat[4];
#pragma unroll
    for (int mt = 0; mt < 4; ++mt)
      amat[mt] = *(const v16bf*)(Kp + (size_t)(mt * 16 + idx) * CN + nbv + half * 16);
#pragma unroll
    for (int et = 0; et < 4; ++et)
      bmat[et] = *(const v16bf*)(Qp + (size_t)(et * 16 + idx) * CN + nbv + half * 16);
#pragma unroll
    for (int mt = 0; mt < 4; ++mt)
#pragma unroll
      for (int et = 0; et < 4; ++et)
        acc[mt][et] = wmma_bf16(amat[mt], bmat[et], acc[mt][et]);
  }

#pragma unroll
  for (int mt = 0; mt < 4; ++mt)
#pragma unroll
    for (int et = 0; et < 4; ++et)
#pragma unroll
      for (int r = 0; r < 8; ++r)
        atomicAdd(&sG[(mt * 16 + half * 8 + r) * 64 + (et * 16 + idx)], acc[mt][et][r]);
  __syncthreads();

  if (tid < 64) {
    const int d = tid;
    const float sd = rk[bh * 64 + d] * rescale[h];
    float mx = -3.4e38f;
    for (int e = 0; e < 64; ++e) {
      float svv = sG[d * 64 + e] * sd * rq[bh * 64 + e];
      mx = fmaxf(mx, svv);
    }
    float sum = 0.0f;
    for (int e = 0; e < 64; ++e)
      sum += __expf(sG[d * 64 + e] * sd * rq[bh * 64 + e] - mx);
    const float inv = 1.0f / sum;
    for (int e = 0; e < 64; ++e)
      attnW[((size_t)bh * 64 + d) * 64 + e] =
          (bf16_t)(__expf(sG[d * 64 + e] * sd * rq[bh * 64 + e] - mx) * inv);
  }
}

// ---------------------------------------------------------------------------
// Kernel 4: xo = attn(64x64) * V(64xn).  The 8KB attention matrix is staged
// once per block into LDS (async-load-to-LDS when available) and shared by all
// 8 waves; V read via v_row for contiguous B lane loads.  Output bf16 in flat
// raw-reshape order.
// ---------------------------------------------------------------------------
__global__ __launch_bounds__(256) void av_kernel(
    const bf16_t* __restrict__ attnW, const bf16_t* __restrict__ vrow,
    bf16_t* __restrict__ xo)
{
  const int bh = blockIdx.y;
  const int b = bh >> 4, h = bh & 15;
  const int tid = threadIdx.x;
  const int wave = tid >> 5, lane = tid & 31, idx = lane & 15, half = lane >> 4;
  const int token = blockIdx.x * 128 + wave * 16 + idx;

  __shared__ __align__(32) bf16_t ldsA[4096];
  const bf16_t* Ag = attnW + (size_t)bh * 4096;
#if HAVE_ASYNC_LDS
  async_b128(Ag + tid * 8, &ldsA[tid * 8]);
  async_b128(Ag + 2048 + tid * 8, &ldsA[2048 + tid * 8]);
  wait_async0();
#else
  *(v8bf*)&ldsA[tid * 8] = *(const v8bf*)(Ag + tid * 8);
  *(v8bf*)&ldsA[2048 + tid * 8] = *(const v8bf*)(Ag + 2048 + tid * 8);
#endif
  __syncthreads();

  v8f acc[4];
#pragma unroll
  for (int mt = 0; mt < 4; ++mt) acc[mt] = vzero8();

#pragma unroll
  for (int kc = 0; kc < 64; kc += 32) {
    v16bf bt = *(const v16bf*)(vrow + ((size_t)b * CN + token) * CINNER + h * 64 + kc + half * 16);
#pragma unroll
    for (int mt = 0; mt < 4; ++mt) {
      v16bf at = *(const v16bf*)&ldsA[(mt * 16 + idx) * 64 + kc + half * 16];
      acc[mt] = wmma_bf16(at, bt, acc[mt]);
    }
  }
#pragma unroll
  for (int mt = 0; mt < 4; ++mt)
#pragma unroll
    for (int r = 0; r < 8; ++r) {
      int m = mt * 16 + half * 8 + r;
      xo[(size_t)b * ((size_t)CINNER * CN) + (size_t)(h * 64 + m) * CN + token] = (bf16_t)acc[mt][r];
    }
}

// ---------------------------------------------------------------------------
// Kernel 5: GEMM out[r,o] = sum_c A2[r,c]*Wb[o,c] + bias[o]
// Block tile 128x64, 8 waves x (2x2) WMMA tiles.  The Wb k-panel is staged
// into double-buffered LDS with a software pipeline: TDM tensor_load_to_lds
// (or per-lane async-load-to-LDS) for the NEXT panel is issued before the
// WMMAs of the current panel, overlapping DMA with math.
// ---------------------------------------------------------------------------
__global__ __launch_bounds__(256) void gemm_proj_kernel(
    const bf16_t* __restrict__ A2, const bf16_t* __restrict__ Wb,
    const float* __restrict__ bias, float* __restrict__ outp)
{
  const int tid = threadIdx.x;
  const int wave = tid >> 5, lane = tid & 31, idx = lane & 15, half = lane >> 4;
  const int rowBlock = blockIdx.x * 128, colBlock = blockIdx.y * 64;
  const int mq = wave & 3, nh = wave >> 2;

  __shared__ __align__(32) bf16_t ldsB[2][64 * 32];
  v8f acc[2][2];
#pragma unroll
  for (int mt = 0; mt < 2; ++mt)
#pragma unroll
    for (int nt = 0; nt < 2; ++nt) acc[mt][nt] = vzero8();

  const int no = tid >> 2, ko8 = (tid & 3) * 8;

  // ---- prologue: stage panel 0 into buffer 0 ----
#if HAVE_TDM
  if (tid < 32) {
    tdm_load_panel(Wb, colBlock, 0, &ldsB[0][0]);
    __builtin_amdgcn_s_wait_tensorcnt(0);
  }
#elif HAVE_ASYNC_LDS
  async_b128(Wb + (size_t)(colBlock + no) * CINNER + ko8, &ldsB[0][no * 32 + ko8]);
  wait_async0();
#else
  *(v8bf*)&ldsB[0][no * 32 + ko8] = *(const v8bf*)(Wb + (size_t)(colBlock + no) * CINNER + ko8);
#endif
  __syncthreads();

  for (int kb = 0; kb < CINNER; kb += 32) {
    const int cur = (kb >> 5) & 1, nxt = cur ^ 1;
    const bool more = (kb + 32) < CINNER;

    // ---- issue next panel's DMA before computing ----
    if (more) {
#if HAVE_TDM
      if (tid < 32) tdm_load_panel(Wb, colBlock, kb + 32, &ldsB[nxt][0]);
#elif HAVE_ASYNC_LDS
      async_b128(Wb + (size_t)(colBlock + no) * CINNER + (kb + 32) + ko8,
                 &ldsB[nxt][no * 32 + ko8]);
#endif
      __builtin_prefetch(A2 + (size_t)(rowBlock + mq * 32 + idx) * CINNER + kb + 32, 0, 1);
    }

    // ---- compute on current panel ----
    v16bf bt[2];
#pragma unroll
    for (int nt = 0; nt < 2; ++nt)
      bt[nt] = *(const v16bf*)&ldsB[cur][(nh * 32 + nt * 16 + idx) * 32 + half * 16];
#pragma unroll
    for (int mt = 0; mt < 2; ++mt) {
      v16bf at = *(const v16bf*)(A2 + (size_t)(rowBlock + mq * 32 + mt * 16 + idx) * CINNER +
                                 kb + half * 16);
#pragma unroll
      for (int nt = 0; nt < 2; ++nt) acc[mt][nt] = wmma_bf16(at, bt[nt], acc[mt][nt]);
    }

    // ---- drain DMA, publish next buffer ----
#if HAVE_TDM
    if (more && tid < 32) __builtin_amdgcn_s_wait_tensorcnt(0);
    __syncthreads();
#elif HAVE_ASYNC_LDS
    if (more) wait_async0();
    __syncthreads();
#else
    __syncthreads();
    if (more)
      *(v8bf*)&ldsB[nxt][no * 32 + ko8] =
          *(const v8bf*)(Wb + (size_t)(colBlock + no) * CINNER + (kb + 32) + ko8);
    __syncthreads();
#endif
  }

#pragma unroll
  for (int mt = 0; mt < 2; ++mt)
#pragma unroll
    for (int nt = 0; nt < 2; ++nt)
#pragma unroll
      for (int r = 0; r < 8; ++r) {
        int row = rowBlock + mq * 32 + mt * 16 + half * 8 + r;
        int col = colBlock + nh * 32 + nt * 16 + idx;
        outp[(size_t)row * CDIM + col] = acc[mt][nt][r] + bias[col];
      }
}

// ---------------------------------------------------------------------------
// Kernel 6/7: NHWC depthwise 3x3 conv (float4 over channels); exact-erf GELU
// variant and accumulate-into-output variant.
// ---------------------------------------------------------------------------
__global__ __launch_bounds__(64) void dwconv_gelu_kernel(
    const float* __restrict__ in, const float* __restrict__ w, float* __restrict__ outp)
{
  const int x = blockIdx.x, y = blockIdx.y, b = blockIdx.z;
  const int c4 = threadIdx.x * 4;
  float wl[4][9];
#pragma unroll
  for (int i = 0; i < 4; ++i)
#pragma unroll
    for (int t = 0; t < 9; ++t) wl[i][t] = w[(c4 + i) * 9 + t];
  float a0 = 0.f, a1 = 0.f, a2 = 0.f, a3 = 0.f;
#pragma unroll
  for (int dy = 0; dy < 3; ++dy)
#pragma unroll
    for (int dx = 0; dx < 3; ++dx) {
      int yy = y + dy - 1, xx = x + dx - 1;
      if (yy < 0 || yy >= CH || xx < 0 || xx >= CW) continue;
      float4 vv = *(const float4*)(in + (((size_t)b * CH + yy) * CW + xx) * CDIM + c4);
      int t = dy * 3 + dx;
      a0 = fmaf(vv.x, wl[0][t], a0);
      a1 = fmaf(vv.y, wl[1][t], a1);
      a2 = fmaf(vv.z, wl[2][t], a2);
      a3 = fmaf(vv.w, wl[3][t], a3);
    }
  float4 o;
  o.x = gelu_exact(a0); o.y = gelu_exact(a1); o.z = gelu_exact(a2); o.w = gelu_exact(a3);
  *(float4*)(outp + (((size_t)b * CH + y) * CW + x) * CDIM + c4) = o;
}

__global__ __launch_bounds__(64) void dwconv_add_kernel(
    const float* __restrict__ in, const float* __restrict__ w, float* __restrict__ outp)
{
  const int x = blockIdx.x, y = blockIdx.y, b = blockIdx.z;
  const int c4 = threadIdx.x * 4;
  float wl[4][9];
#pragma unroll
  for (int i = 0; i < 4; ++i)
#pragma unroll
    for (int t = 0; t < 9; ++t) wl[i][t] = w[(c4 + i) * 9 + t];
  float a0 = 0.f, a1 = 0.f, a2 = 0.f, a3 = 0.f;
#pragma unroll
  for (int dy = 0; dy < 3; ++dy)
#pragma unroll
    for (int dx = 0; dx < 3; ++dx) {
      int yy = y + dy - 1, xx = x + dx - 1;
      if (yy < 0 || yy >= CH || xx < 0 || xx >= CW) continue;
      float4 vv = *(const float4*)(in + (((size_t)b * CH + yy) * CW + xx) * CDIM + c4);
      int t = dy * 3 + dx;
      a0 = fmaf(vv.x, wl[0][t], a0);
      a1 = fmaf(vv.y, wl[1][t], a1);
      a2 = fmaf(vv.z, wl[2][t], a2);
      a3 = fmaf(vv.w, wl[3][t], a3);
    }
  size_t off = (((size_t)b * CH + y) * CW + x) * CDIM + c4;
  float4 o = *(float4*)(outp + off);
  o.x += a0; o.y += a1; o.z += a2; o.w += a3;
  *(float4*)(outp + off) = o;
}

// ---------------------------------------------------------------------------
// Host launcher
// ---------------------------------------------------------------------------
extern "C" void kernel_launch(void* const* d_in, const int* in_sizes, int n_in,
                              void* d_out, int out_size, void* d_ws, size_t ws_size,
                              hipStream_t stream) {
  const float* x_in   = (const float*)d_in[0];
  const float* wq     = (const float*)d_in[1];
  const float* wk     = (const float*)d_in[2];
  const float* wv     = (const float*)d_in[3];
  const float* bnq_g  = (const float*)d_in[4];
  const float* bnq_b  = (const float*)d_in[5];
  const float* bnq_m  = (const float*)d_in[6];
  const float* bnq_v  = (const float*)d_in[7];
  const float* bnk_g  = (const float*)d_in[8];
  const float* bnk_b  = (const float*)d_in[9];
  const float* bnk_m  = (const float*)d_in[10];
  const float* bnk_v  = (const float*)d_in[11];
  const float* bnv_g  = (const float*)d_in[12];
  const float* bnv_b  = (const float*)d_in[13];
  const float* bnv_m  = (const float*)d_in[14];
  const float* bnv_v  = (const float*)d_in[15];
  const float* rescale= (const float*)d_in[16];
  const float* proj_w = (const float*)d_in[17];
  const float* proj_b = (const float*)d_in[18];
  const float* pos_w1 = (const float*)d_in[19];
  const float* pos_w2 = (const float*)d_in[20];
  (void)in_sizes; (void)n_in; (void)out_size; (void)ws_size;

  char* ws = (char*)d_ws;
  // workspace layout (bytes)
  bf16_t* qh    = (bf16_t*)(ws + 0);              // 64 MiB [b, co, n]
  bf16_t* kh    = (bf16_t*)(ws + 67108864);       // 64 MiB
  bf16_t* vrow  = (bf16_t*)(ws + 134217728);      // 64 MiB [b, n, 1024]
  bf16_t* xo    = (bf16_t*)(ws + 201326592);      // 64 MiB flat raw-reshape layout
  float*  vp    = (float* )(ws + 268435456);      // 32 MiB v_inp @ W^T + b (NHWC)
  float*  t1    = (float* )(ws + 301989888);      // 32 MiB gelu(dwconv(vp))
  bf16_t* attnW = (bf16_t*)(ws + 335544320);      // 256 KiB
  float*  rq    = (float* )(ws + 335806464);      // 8 KiB
  float*  rk    = (float* )(ws + 335814656);      // 8 KiB
  bf16_t* wpb   = (bf16_t*)(ws + 335822848);      // 512 KiB bf16 proj_w
  float*  outp  = (float*)d_out;

  // 0) proj_w -> bf16
  cvt_bf16_kernel<<<dim3(1024), dim3(256), 0, stream>>>(proj_w, wpb, CDIM * CINNER);

  // 1) grouped conv + BN -> qh, kh, vrow
  qkv_conv_bn_kernel<<<dim3(CW, CH, CB), dim3(256), 0, stream>>>(
      x_in, wq, wk, wv,
      bnq_g, bnq_b, bnq_m, bnq_v,
      bnk_g, bnk_b, bnk_m, bnk_v,
      bnv_g, bnv_b, bnv_m, bnv_v,
      qh, kh, vrow);

  // 2) reciprocal row norms of q and k
  rownorm_kernel<<<dim3(CB * CINNER), dim3(256), 0, stream>>>(qh, rq);
  rownorm_kernel<<<dim3(CB * CINNER), dim3(256), 0, stream>>>(kh, rk);

  // 3) logits + softmax per (b,head)
  attn_kernel<<<dim3(CB * CHEADS), dim3(256), 0, stream>>>(qh, kh, rq, rk, rescale, attnW);

  // 4) xo = attn @ V
  av_kernel<<<dim3(CN / 128, CB * CHEADS), dim3(256), 0, stream>>>(attnW, vrow, xo);

  // 5) projections: out_c -> d_out ; vp -> ws
  gemm_proj_kernel<<<dim3((CB * CN) / 128, CDIM / 64), dim3(256), 0, stream>>>(xo, wpb, proj_b, outp);
  gemm_proj_kernel<<<dim3((CB * CN) / 128, CDIM / 64), dim3(256), 0, stream>>>(vrow, wpb, proj_b, vp);

  // 6) positional path: t1 = gelu(dwconv(vp, w1)); d_out += dwconv(t1, w2)
  dwconv_gelu_kernel<<<dim3(CW, CH, CB), dim3(64), 0, stream>>>(vp, pos_w1, t1);
  dwconv_add_kernel<<<dim3(CW, CH, CB), dim3(64), 0, stream>>>(t1, pos_w2, outp);
}
